// LorentzGraphDecoder_27685359190110
// MI455X (gfx1250) — compile-verified, matrix operand order checked by reading, not run on hardware
//
#include <hip/hip_runtime.h>
#include <hip/hip_bf16.h>
#include <math.h>

typedef __attribute__((ext_vector_type(16))) _Float16 v16h;
typedef __attribute__((ext_vector_type(8)))  _Float16 v8h;
typedef __attribute__((ext_vector_type(8)))  float    v8f;
typedef __attribute__((ext_vector_type(4)))  float    v4f;

#define N_ROWS 8192
#define D      64
#define PADC   80               // 65 columns of h padded to 5 WMMA N-tiles
#define KSPLIT 2                // K-range split for occupancy (1024 waves)
#define KRANGE (N_ROWS / KSPLIT)

// ---------------------------------------------------------------------------
// Kernel 1: fused logmap0 -> (u @ W^T) -> expmap0.
// Writes h transposed as f16: hT[c][row], c in [0,80), cols 65..79 zeroed.
// Transposed layout makes each WMMA B-operand column a contiguous K-run.
// ---------------------------------------------------------------------------
__global__ void lgd_front(const float* __restrict__ x,
                          const float* __restrict__ W,
                          _Float16* __restrict__ hT) {
  __shared__ float Ws[D * D];
  for (int i = threadIdx.x; i < D * D; i += blockDim.x) Ws[i] = W[i];
  __syncthreads();

  int row = blockIdx.x * blockDim.x + threadIdx.x;
  if (row >= N_ROWS) return;
  const float* xr = x + (size_t)row * (D + 1);

  // logmap0_tail: u = acosh(max(x0, 1+eps)) * y / max(||y||, 1e-15)
  float y[D]; float yn2 = 0.f;
  #pragma unroll
  for (int t = 0; t < D; ++t) { float v = xr[1 + t]; y[t] = v; yn2 = fmaf(v, v, yn2); }
  float ynorm = fmaxf(sqrtf(yn2), 1e-15f);
  float theta = acoshf(fmaxf(xr[0], 1.0f + 1e-7f));
  float scale = theta / ynorm;

  // mu = u @ W^T  (broadcast LDS reads, uniform address per wave)
  float mu[D]; float mn2 = 0.f;
  for (int c = 0; c < D; ++c) {
    float acc = 0.f;
    #pragma unroll 8
    for (int k = 0; k < D; ++k) acc = fmaf(y[k], Ws[c * D + k], acc);
    acc *= scale;
    mu[c] = acc; mn2 = fmaf(acc, acc, mn2);
  }

  // expmap0: h0 = cosh(||mu||), ht = sinh(||mu||) * mu / ||mu||
  float mnorm = fmaxf(sqrtf(mn2), 1e-15f);
  float ch = coshf(mnorm);
  float sh = sinhf(mnorm) / mnorm;

  hT[row] = (_Float16)ch;
  #pragma unroll
  for (int c = 0; c < D; ++c)
    hT[(size_t)(c + 1) * N_ROWS + row] = (_Float16)(sh * mu[c]);
  #pragma unroll
  for (int c = D + 1; c < PADC; ++c)   // zero pad cols every call (ws is poisoned)
    hT[(size_t)c * N_ROWS + row] = (_Float16)0.f;
}

// ---------------------------------------------------------------------------
// Kernel 2: s = adj @ h via v_wmma_f32_16x16x32_f16, software-pipelined,
// K-split x2. One wave = one 16-row strip x 5 N-tiles over half the K range;
// partial sums land in s_part[half]. 1024 waves total (~2 per SIMD32) so
// wave-level interleave + intra-wave lookahead keep HBM requests in flight.
// adj streamed once from HBM with NT hint (fp32 -> f16 in regs); hT L2-resident.
// ---------------------------------------------------------------------------
__global__ void lgd_gemm(const float* __restrict__ adj,
                         const _Float16* __restrict__ hT,
                         float* __restrict__ s) {
  const int lane  = threadIdx.x & 31;
  const int wid   = (blockIdx.x * blockDim.x + threadIdx.x) >> 5;  // 0..1023
  const int strip = wid >> 1;               // 16-row strip, 0..511
  const int half  = wid & 1;                // K-range half
  const int kbase = half * KRANGE;
  const int r     = lane & 15;
  const int hs    = lane >> 4;              // half-select (K split per ISA 7.12.2)

  const float*    Arow  = adj + (size_t)(strip * 16 + r) * N_ROWS + kbase;
  const _Float16* Bbase = hT + (size_t)r * N_ROWS + kbase + hs * 8;

  v8f acc[5] = {};  // 5 accumulators: 16x16 f32 tiles, 8 VGPRs each

  // in-flight register set (load targets)
  v4f A0, A1, A2, A3;
  v8h B0[5], B1[5];

  auto load_tile = [&](int kk) {
    const v4f* p0 = (const v4f*)(Arow + kk + hs * 8);
    const v4f* p1 = (const v4f*)(Arow + kk + 16 + hs * 8);
    A0 = __builtin_nontemporal_load(p0);       // adj: streamed once -> NT
    A1 = __builtin_nontemporal_load(p0 + 1);
    A2 = __builtin_nontemporal_load(p1);
    A3 = __builtin_nontemporal_load(p1 + 1);
    #pragma unroll
    for (int c = 0; c < 5; ++c) {              // B tiles: c*16*8192*2B immediate offsets
      const _Float16* pb = Bbase + (size_t)(c * 16) * N_ROWS + kk;
      B0[c] = *(const v8h*)(pb);
      B1[c] = *(const v8h*)(pb + 16);
    }
  };

  load_tile(0);

  for (int k = 0; k < KRANGE; k += 32) {
    // Drain in-flight set into WMMA operands (waits land here, one iter late)
    v16h a;
    a[0]  = (_Float16)A0[0]; a[1]  = (_Float16)A0[1]; a[2]  = (_Float16)A0[2]; a[3]  = (_Float16)A0[3];
    a[4]  = (_Float16)A1[0]; a[5]  = (_Float16)A1[1]; a[6]  = (_Float16)A1[2]; a[7]  = (_Float16)A1[3];
    a[8]  = (_Float16)A2[0]; a[9]  = (_Float16)A2[1]; a[10] = (_Float16)A2[2]; a[11] = (_Float16)A2[3];
    a[12] = (_Float16)A3[0]; a[13] = (_Float16)A3[1]; a[14] = (_Float16)A3[2]; a[15] = (_Float16)A3[3];
    v16h b[5];
    #pragma unroll
    for (int c = 0; c < 5; ++c) {
      #pragma unroll
      for (int i = 0; i < 8; ++i) { b[c][i] = B0[c][i]; b[c][8 + i] = B1[c][i]; }
    }

    // Issue next iteration's 14 loads (wrap keeps the last prefetch in-bounds)
    load_tile((k + 32) & (KRANGE - 1));

    #pragma unroll
    for (int c = 0; c < 5; ++c)
      acc[c] = __builtin_amdgcn_wmma_f32_16x16x32_f16(
          /*neg_a=*/false, a, /*neg_b=*/false, b[c],
          /*c_mod=*/(short)0, acc[c], /*reuse_a=*/false, /*reuse_b=*/false);
  }

  // C/D layout: VGPR v -> M = v + hs*8, lane r -> N.  Partial buffer per half.
  float* sp = s + (size_t)half * N_ROWS * PADC;
  #pragma unroll
  for (int c = 0; c < 5; ++c) {
    #pragma unroll
    for (int v = 0; v < 8; ++v)
      sp[(size_t)(strip * 16 + hs * 8 + v) * PADC + c * 16 + r] = acc[c][v];
  }
}

// ---------------------------------------------------------------------------
// Kernel 3: combine K-split partials, then Lorentz normalize ->
// relu(logmap0) -> expmap0 -> Poincare map.
// ---------------------------------------------------------------------------
__global__ void lgd_back(const float* __restrict__ s, float* __restrict__ out) {
  int row = blockIdx.x * blockDim.x + threadIdx.x;
  if (row >= N_ROWS) return;
  const float* sr0 = s + (size_t)row * PADC;
  const float* sr1 = sr0 + (size_t)N_ROWS * PADC;

  float s0 = sr0[0] + sr1[0];
  float st[D];
  float inner = -s0 * s0;
  #pragma unroll
  for (int t = 0; t < D; ++t) {
    float v = sr0[1 + t] + sr1[1 + t];
    st[t] = v; inner = fmaf(v, v, inner);
  }

  float coef = 1.0f / sqrtf(fmaxf(fabsf(inner), 1e-15f));
  float h0 = coef * s0;

  float yn2 = 0.f;
  #pragma unroll
  for (int t = 0; t < D; ++t) { float v = coef * st[t]; st[t] = v; yn2 = fmaf(v, v, yn2); }
  float ynorm = fmaxf(sqrtf(yn2), 1e-15f);
  float th = acoshf(fmaxf(h0, 1.0f + 1e-7f));
  float sc = th / ynorm;

  float hn2 = 0.f;
  #pragma unroll
  for (int t = 0; t < D; ++t) { float v = fmaxf(sc * st[t], 0.f); st[t] = v; hn2 = fmaf(v, v, hn2); }
  float hn = fmaxf(sqrtf(hn2), 1e-15f);
  float ch = coshf(hn);
  float sh = sinhf(hn) / hn;
  float inv = 1.0f / (ch + 1.0f);   // SQRT_K = 1

  #pragma unroll
  for (int t = 0; t < D; ++t) out[(size_t)row * D + t] = sh * st[t] * inv;
}

// ---------------------------------------------------------------------------
extern "C" void kernel_launch(void* const* d_in, const int* in_sizes, int n_in,
                              void* d_out, int out_size, void* d_ws, size_t ws_size,
                              hipStream_t stream) {
  const float* x   = (const float*)d_in[0];   // (8192, 65)
  const float* adj = (const float*)d_in[1];   // (8192, 8192)
  const float* W   = (const float*)d_in[2];   // (64, 64)
  // d_in[3] = dec_bias: unused by the reference computation
  float* out = (float*)d_out;

  _Float16* hT = (_Float16*)d_ws;                                   // 80*8192*2 = 1.25 MB
  float*    s  = (float*)((char*)d_ws + (size_t)PADC * N_ROWS * 2); // 2 * 80*8192*4 = 5 MB

  lgd_front<<<N_ROWS / 256, 256, 0, stream>>>(x, W, hT);
  lgd_gemm<<<128, 256, 0, stream>>>(adj, hT, s);   // 1024 waves = 512 strips x 2 K-halves
  lgd_back<<<N_ROWS / 256, 256, 0, stream>>>(s, out);

  // Second tuple output: adj passed through unchanged.
  hipMemcpyAsync(out + (size_t)N_ROWS * D, adj,
                 (size_t)N_ROWS * N_ROWS * sizeof(float),
                 hipMemcpyDeviceToDevice, stream);
}
